// S2VTModel_2336462209329
// MI455X (gfx1250) — matrix operand order repaired and checked by reading, not compiled, not run on hardware
//
#include <hip/hip_runtime.h>

typedef __bf16 bf16_t;
typedef __attribute__((ext_vector_type(16))) __bf16 v16bf;
typedef __attribute__((ext_vector_type(8)))  float  v8f;
typedef __attribute__((ext_vector_type(4)))  unsigned int v4u;

namespace {
constexpr int kB   = 64;     // batch
constexpr int kT   = 40;     // encoder steps
constexpr int kDV  = 2048;   // video feature dim
constexpr int kDH  = 512;    // hidden dim
constexpr int kDW  = 512;    // word dim
constexpr int kG3  = 3 * kDH;          // 1536 gate columns
constexpr int kDEC = 27;               // MAX_LEN - 1 decode steps
constexpr int kML  = 28;               // MAX_LEN
constexpr int kV   = 16000;            // vocab
}

#define WMMA_BF16(a, b, c) \
  __builtin_amdgcn_wmma_f32_16x16x32_bf16(false, (a), false, (b), (short)0, (c), false, false)

// A fragment (16x32 bf16, row-major source): lane<16 -> K [k0..k0+15],
// lane>=16 -> K [k0+16..k0+31]. Two 16B loads.
__device__ __forceinline__ v16bf load_a_frag(const bf16_t* __restrict__ row, int k0, int koff8) {
  union { v16bf v; v4u u[2]; } f;
  f.u[0] = *(const v4u*)(row + k0 + 2 * koff8);
  f.u[1] = *(const v4u*)(row + k0 + 2 * koff8 + 8);
  return f.v;
}
// B fragment (32x16 bf16; source stored as row n = weight row, contiguous in K):
// lane<16 -> K [k0..k0+15], lane>=16 -> K [k0+16..k0+31]. Two 16B loads.
__device__ __forceinline__ v16bf load_b_frag(const bf16_t* __restrict__ row, int k0, int koff16) {
  union { v16bf v; v4u u[2]; } f;
  f.u[0] = *(const v4u*)(row + k0 + koff16);
  f.u[1] = *(const v4u*)(row + k0 + koff16 + 8);
  return f.v;
}

// ---------------- setup / conversion kernels ----------------

__global__ void k_zero_u32(unsigned int* __restrict__ p, int n) {
  int i = blockIdx.x * blockDim.x + threadIdx.x;
  if (i < n) p[i] = 0u;
}

__global__ void k_cvt_bf16(const float* __restrict__ s, bf16_t* __restrict__ d, int n) {
  int i = blockIdx.x * blockDim.x + threadIdx.x;
  if (i < n) d[i] = (bf16_t)s[i];
}

// vid_feats (B,T,DV) fp32 -> (T,B,DV) bf16
__global__ void k_cvt_vid(const float* __restrict__ vid, bf16_t* __restrict__ d) {
  int i = blockIdx.x * blockDim.x + threadIdx.x;
  if (i >= kB * kT * kDV) return;
  int k = i % kDV;
  int t = (i / kDV) % kT;
  int b = i / (kDV * kT);
  d[((size_t)t * kB + b) * kDV + k] = (bf16_t)vid[i];
}

// words[t][b][:] = bf16(emb[target[b][t]][:]) for t in [0,27)
__global__ void k_gather_words(const int* __restrict__ tgt, const float* __restrict__ emb,
                               bf16_t* __restrict__ d) {
  int i = blockIdx.x * blockDim.x + threadIdx.x;
  if (i >= kDEC * kB * kDH) return;
  int k = i % kDH;
  int b = (i / kDH) % kB;
  int t = i / (kDH * kB);
  int idx = tgt[b * kML + t];
  d[i] = (bf16_t)emb[(size_t)idx * kDH + k];
}

// ---------------- big WMMA GEMM: C[MxN] = A[MxK] * B[NxK]^T ----------------
// grid.x = N/64, grid.y = M/64, block = 128 (4 waves, one 16-row M-tile each).
// Each wave computes 4 N-tiles, reusing one A fragment across 4 WMMAs
// (2.5 b128 loads per WMMA instead of 4).
__global__ void __launch_bounds__(128)
k_gemm_bf16(const bf16_t* __restrict__ A, int lda,
            const bf16_t* __restrict__ B, int ldb,
            float* __restrict__ C, int ldc, int K) {
  const int lane = threadIdx.x & 31;
  const int wave = threadIdx.x >> 5;
  const int half = lane >> 4;
  const int lidx = lane & 15;
  const int koff8 = half * 8, koff16 = half * 16;
  const int n0 = blockIdx.x * 64;
  const int m0 = blockIdx.y * 64 + wave * 16;

  const bf16_t* ap  = A + (size_t)(m0 + lidx) * lda;
  const bf16_t* bp0 = B + (size_t)(n0 +  0 + lidx) * ldb;
  const bf16_t* bp1 = B + (size_t)(n0 + 16 + lidx) * ldb;
  const bf16_t* bp2 = B + (size_t)(n0 + 32 + lidx) * ldb;
  const bf16_t* bp3 = B + (size_t)(n0 + 48 + lidx) * ldb;

  v8f acc0 = {}, acc1 = {}, acc2 = {}, acc3 = {};
  for (int k0 = 0; k0 < K; k0 += 32) {
    v16bf a = load_a_frag(ap, k0, koff8);
    acc0 = WMMA_BF16(a, load_b_frag(bp0, k0, koff16), acc0);
    acc1 = WMMA_BF16(a, load_b_frag(bp1, k0, koff16), acc1);
    acc2 = WMMA_BF16(a, load_b_frag(bp2, k0, koff16), acc2);
    acc3 = WMMA_BF16(a, load_b_frag(bp3, k0, koff16), acc3);
  }
  const int c = n0 + lidx;
#pragma unroll
  for (int e = 0; e < 8; ++e) {
    const int row = m0 + e + half * 8;
    float* crow = C + (size_t)row * ldc;
    crow[c +  0] = acc0[e];
    crow[c + 16] = acc1[e];
    crow[c + 32] = acc2[e];
    crow[c + 48] = acc3[e];
  }
}

// ---------------- fused GRU step ----------------
// grid = 32 (DH/16 column slabs), block = 128 (4 waves = 64 batch rows).
// Gates: optional precomputed input-GEMM (gi_pre), optional in-step input GEMM
// (Ain, Kx>0), always the hidden GEMM over K=512. Fused sigmoid/tanh epilogue
// writes new h in fp32 + bf16.
__global__ void __launch_bounds__(128)
k_gru_step(const float* __restrict__ gi_pre,          // [64][1536] or null
           const bf16_t* __restrict__ Ain, int lda, int Kx,
           const bf16_t* __restrict__ Wih, int ldw,   // rows: [r|z|n] x Kx
           const float* __restrict__ bih,
           const bf16_t* __restrict__ Whh,            // [1536][512]
           const float* __restrict__ bhh,
           const bf16_t* __restrict__ Hprevb,         // [64][512] bf16
           const float* __restrict__ Hprevf,          // [64][512] fp32
           float* __restrict__ Houtf,
           bf16_t* __restrict__ Houtb) {
  const int lane = threadIdx.x & 31;
  const int wave = threadIdx.x >> 5;
  const int half = lane >> 4;
  const int lidx = lane & 15;
  const int koff8 = half * 8, koff16 = half * 16;
  const int j = blockIdx.x * 16;                      // h-column slab

  v8f ir_a = {}, iz_a = {}, in_a = {};
  v8f hr_a = {}, hz_a = {}, hn_a = {};

  if (Kx > 0) {
    const bf16_t* ap  = Ain + (size_t)(wave * 16 + lidx) * lda;
    const bf16_t* bp0 = Wih + (size_t)(0 * kDH + j + lidx) * ldw;
    const bf16_t* bp1 = Wih + (size_t)(1 * kDH + j + lidx) * ldw;
    const bf16_t* bp2 = Wih + (size_t)(2 * kDH + j + lidx) * ldw;
    for (int k0 = 0; k0 < Kx; k0 += 32) {
      v16bf a = load_a_frag(ap, k0, koff8);
      ir_a = WMMA_BF16(a, load_b_frag(bp0, k0, koff16), ir_a);
      iz_a = WMMA_BF16(a, load_b_frag(bp1, k0, koff16), iz_a);
      in_a = WMMA_BF16(a, load_b_frag(bp2, k0, koff16), in_a);
    }
  }
  {
    const bf16_t* ap  = Hprevb + (size_t)(wave * 16 + lidx) * kDH;
    const bf16_t* bp0 = Whh + (size_t)(0 * kDH + j + lidx) * kDH;
    const bf16_t* bp1 = Whh + (size_t)(1 * kDH + j + lidx) * kDH;
    const bf16_t* bp2 = Whh + (size_t)(2 * kDH + j + lidx) * kDH;
    for (int k0 = 0; k0 < kDH; k0 += 32) {
      v16bf a = load_a_frag(ap, k0, koff8);
      hr_a = WMMA_BF16(a, load_b_frag(bp0, k0, koff16), hr_a);
      hz_a = WMMA_BF16(a, load_b_frag(bp1, k0, koff16), hz_a);
      hn_a = WMMA_BF16(a, load_b_frag(bp2, k0, koff16), hn_a);
    }
  }

  const int c = j + lidx;
  const float bir = bih[c], biz = bih[kDH + c], bin = bih[2 * kDH + c];
  const float bhr = bhh[c], bhz = bhh[kDH + c], bhn = bhh[2 * kDH + c];
#pragma unroll
  for (int e = 0; e < 8; ++e) {
    const int row = wave * 16 + e + half * 8;       // batch index 0..63
    float gi_r = ir_a[e] + bir;
    float gi_z = iz_a[e] + biz;
    float gi_n = in_a[e] + bin;
    if (gi_pre) {
      const float* g = gi_pre + (size_t)row * kG3;
      gi_r += g[c];
      gi_z += g[kDH + c];
      gi_n += g[2 * kDH + c];
    }
    const float gh_r = hr_a[e] + bhr;
    const float gh_z = hz_a[e] + bhz;
    const float gh_n = hn_a[e] + bhn;
    const float r = 1.f / (1.f + expf(-(gi_r + gh_r)));
    const float z = 1.f / (1.f + expf(-(gi_z + gh_z)));
    const float n = tanhf(gi_n + r * gh_n);
    const float hold = Hprevf[(size_t)row * kDH + c];
    const float hnew = (1.f - z) * n + z * hold;
    Houtf[(size_t)row * kDH + c] = hnew;
    Houtb[(size_t)row * kDH + c] = (bf16_t)hnew;
  }
}

// ---------------- bias + log_softmax over vocab, write to d_out ----------------
__global__ void __launch_bounds__(256)
k_logsoftmax(const float* __restrict__ logits, const float* __restrict__ bout,
             float* __restrict__ out, int t) {
  const int b = blockIdx.x;
  const int tid = threadIdx.x;
  const float* row = logits + (size_t)b * kV;
  __shared__ float red[256];

  float m = -1e30f;
  for (int i = tid; i < kV; i += 256) m = fmaxf(m, row[i] + bout[i]);
  red[tid] = m;
  __syncthreads();
  for (int s = 128; s > 0; s >>= 1) {
    if (tid < s) red[tid] = fmaxf(red[tid], red[tid + s]);
    __syncthreads();
  }
  m = red[0];
  __syncthreads();

  float sum = 0.f;
  for (int i = tid; i < kV; i += 256) sum += expf(row[i] + bout[i] - m);
  red[tid] = sum;
  __syncthreads();
  for (int s = 128; s > 0; s >>= 1) {
    if (tid < s) red[tid] += red[tid + s];
    __syncthreads();
  }
  const float denom = m + logf(red[0]);

  float* orow = out + (size_t)b * (kDEC * kV) + (size_t)t * kV;
  for (int i = tid; i < kV; i += 256) orow[i] = row[i] + bout[i] - denom;
}

// ---------------- host driver ----------------

extern "C" void kernel_launch(void* const* d_in, const int* in_sizes, int n_in,
                              void* d_out, int out_size, void* d_ws, size_t ws_size,
                              hipStream_t stream) {
  (void)in_sizes; (void)n_in; (void)out_size; (void)ws_size;
  const float* vid   = (const float*)d_in[0];
  const int*   tgt   = (const int*)d_in[1];
  const float* w_ih1 = (const float*)d_in[2];
  const float* w_hh1 = (const float*)d_in[3];
  const float* b_ih1 = (const float*)d_in[4];
  const float* b_hh1 = (const float*)d_in[5];
  const float* w_ih2 = (const float*)d_in[6];
  const float* w_hh2 = (const float*)d_in[7];
  const float* b_ih2 = (const float*)d_in[8];
  const float* b_hh2 = (const float*)d_in[9];
  const float* emb   = (const float*)d_in[10];
  const float* w_out = (const float*)d_in[11];
  const float* b_out = (const float*)d_in[12];
  float* out = (float*)d_out;

  char* base = (char*)d_ws;
  size_t off = 0;
  auto alloc = [&](size_t bytes) -> char* {
    char* p = base + off;
    off = (off + bytes + 255) & ~(size_t)255;
    return p;
  };

  // bf16 weight / activation mirrors
  bf16_t* wih1b  = (bf16_t*)alloc((size_t)kG3 * kDV * 2);
  bf16_t* whh1b  = (bf16_t*)alloc((size_t)kG3 * kDH * 2);
  bf16_t* wih2b  = (bf16_t*)alloc((size_t)kG3 * (kDH + kDW) * 2);
  bf16_t* whh2b  = (bf16_t*)alloc((size_t)kG3 * kDH * 2);
  bf16_t* woutb  = (bf16_t*)alloc((size_t)kV * kDH * 2);
  bf16_t* vidb   = (bf16_t*)alloc((size_t)kT * kB * kDV * 2);
  bf16_t* wordsb = (bf16_t*)alloc((size_t)kDEC * kB * kDH * 2);
  // precomputed input-gate GEMMs
  float* gi1  = (float*)alloc((size_t)kT * kB * kG3 * 4);    // enc: x_t @ Wih1^T
  float* gi2w = (float*)alloc((size_t)kDEC * kB * kG3 * 4);  // dec: w_t @ Wih2[:,512:]^T
  float* logits = (float*)alloc((size_t)kB * kV * 4);
  // zero-init region: hidden state ping-pong buffers (fp32 + bf16)
  char* zbeg = base + off;
  float*  h1f[2]; float*  h2f[2]; bf16_t* h1b[2]; bf16_t* h2b[2];
  for (int i = 0; i < 2; ++i) { h1f[i] = (float*)alloc((size_t)kB * kDH * 4);
                                h2f[i] = (float*)alloc((size_t)kB * kDH * 4); }
  for (int i = 0; i < 2; ++i) { h1b[i] = (bf16_t*)alloc((size_t)kB * kDH * 2);
                                h2b[i] = (bf16_t*)alloc((size_t)kB * kDH * 2); }
  const int zn = (int)(((base + off) - zbeg) / 4);
  k_zero_u32<<<(zn + 255) / 256, 256, 0, stream>>>((unsigned int*)zbeg, zn);

  // fp32 -> bf16 conversions
  auto cvt = [&](const float* s, bf16_t* d, int n) {
    k_cvt_bf16<<<(n + 255) / 256, 256, 0, stream>>>(s, d, n);
  };
  cvt(w_ih1, wih1b, kG3 * kDV);
  cvt(w_hh1, whh1b, kG3 * kDH);
  cvt(w_ih2, wih2b, kG3 * (kDH + kDW));
  cvt(w_hh2, whh2b, kG3 * kDH);
  cvt(w_out, woutb, kV * kDH);
  k_cvt_vid<<<(kB * kT * kDV + 255) / 256, 256, 0, stream>>>(vid, vidb);
  k_gather_words<<<(kDEC * kB * kDH + 255) / 256, 256, 0, stream>>>(tgt, emb, wordsb);

  // Hoisted input GEMMs (big, parallel):
  // gi1[T*B, 1536] = vidb[T*B, 2048] @ wih1b^T
  k_gemm_bf16<<<dim3(kG3 / 64, (kT * kB) / 64), 128, 0, stream>>>(
      vidb, kDV, wih1b, kDV, gi1, kG3, kDV);
  // gi2w[DEC*B, 1536] = wordsb[DEC*B, 512] @ wih2b[:, 512:1024]^T
  k_gemm_bf16<<<dim3(kG3 / 64, (kDEC * kB) / 64), 128, 0, stream>>>(
      wordsb, kDH, wih2b + kDH, kDH + kDW, gi2w, kG3, kDH);

  int c1 = 0, c2 = 0;
  // -------- encoder scan --------
  for (int t = 0; t < kT; ++t) {
    // GRU1: input gates precomputed; only hidden GEMM in-step
    k_gru_step<<<kDH / 16, 128, 0, stream>>>(
        gi1 + (size_t)t * kB * kG3, (const bf16_t*)nullptr, 0, 0,
        whh1b, kDH, b_ih1, whh1b, b_hh1,
        h1b[c1], h1f[c1], h1f[c1 ^ 1], h1b[c1 ^ 1]);
    c1 ^= 1;
    // GRU2: in2 = [h1 | 0]; word half contributes nothing -> Kx=512 over h1 only
    k_gru_step<<<kDH / 16, 128, 0, stream>>>(
        (const float*)nullptr, h1b[c1], kDH, kDH,
        wih2b, kDH + kDW, b_ih2, whh2b, b_hh2,
        h2b[c2], h2f[c2], h2f[c2 ^ 1], h2b[c2 ^ 1]);
    c2 ^= 1;
  }
  // -------- decoder scan --------
  for (int t = 0; t < kDEC; ++t) {
    // GRU1: zero input -> gates are bias + hidden GEMM only
    k_gru_step<<<kDH / 16, 128, 0, stream>>>(
        (const float*)nullptr, (const bf16_t*)nullptr, 0, 0,
        whh1b, kDH, b_ih1, whh1b, b_hh1,
        h1b[c1], h1f[c1], h1f[c1 ^ 1], h1b[c1 ^ 1]);
    c1 ^= 1;
    // GRU2: in2 = [h1 | w_t]; word half precomputed in gi2w
    k_gru_step<<<kDH / 16, 128, 0, stream>>>(
        gi2w + (size_t)t * kB * kG3, h1b[c1], kDH, kDH,
        wih2b, kDH + kDW, b_ih2, whh2b, b_hh2,
        h2b[c2], h2f[c2], h2f[c2 ^ 1], h2b[c2 ^ 1]);
    c2 ^= 1;
    // logits = h2 @ w_out^T  (bias folded into log_softmax)
    k_gemm_bf16<<<dim3(kV / 64, 1), 128, 0, stream>>>(
        h2b[c2], kDH, woutb, kDH, logits, kV, kDH);
    k_logsoftmax<<<kB, 256, 0, stream>>>(logits, b_out, out, t);
  }
}
